// GCN_32779190403462
// MI455X (gfx1250) — matrix operand order repaired and checked by reading, not compiled, not run on hardware
//
#include <hip/hip_runtime.h>

typedef __attribute__((ext_vector_type(16))) __bf16 v16bf;
typedef __attribute__((ext_vector_type(8)))  float  v8f;
typedef unsigned short u16;

union V16 {           // 32-byte WMMA fragment, loaded as two b128s
  v16bf v;
  uint4 u[2];
};

__device__ __forceinline__ u16 f2bf(float f) {   // RNE float -> bf16
  unsigned int u = __float_as_uint(f);
  unsigned int r = u + 0x7FFFu + ((u >> 16) & 1u);
  return (u16)(r >> 16);
}
__device__ __forceinline__ unsigned int f2ord(float f) {  // order-preserving f32->u32
  unsigned int u = __float_as_uint(f);
  return (u & 0x80000000u) ? ~u : (u | 0x80000000u);
}
__device__ __forceinline__ float ord2f(unsigned int u) {
  return __uint_as_float((u & 0x80000000u) ? (u & 0x7FFFFFFFu) : ~u);
}

// ---------------------------------------------------------------- cast x->bf16
__global__ void k_cast_bf16(const float* __restrict__ in, u16* __restrict__ out,
                            long long count4) {
  long long i = (long long)blockIdx.x * blockDim.x + threadIdx.x;
  const long long stride = (long long)gridDim.x * blockDim.x;
  for (; i < count4; i += stride) {
    float4 f = ((const float4*)in)[i];
    unsigned int lo = (unsigned int)f2bf(f.x) | ((unsigned int)f2bf(f.y) << 16);
    unsigned int hi = (unsigned int)f2bf(f.z) | ((unsigned int)f2bf(f.w) << 16);
    uint2 v; v.x = lo; v.y = hi;
    ((uint2*)out)[i] = v;
  }
}

// ------------------------------------------- repack W[K,N] -> Wp[K/16][N][16] bf16
__global__ void k_repack_w(const float* __restrict__ W, u16* __restrict__ Wp,
                           int K, int Nout) {
  int tid = blockIdx.x * blockDim.x + threadIdx.x;
  int total = (K >> 4) * Nout;
  if (tid >= total) return;
  int kb = tid / Nout, n = tid - kb * Nout;
#pragma unroll
  for (int i = 0; i < 16; ++i)
    Wp[(size_t)tid * 16 + i] = f2bf(W[(size_t)(kb * 16 + i) * Nout + n]);
}

// ------------------------------------------------------------ WMMA bf16 GEMM
// A: [M,K] bf16 row-major; Wp packed. Wave computes 32 rows x 64 cols with
// register double-buffering of the next k-slice (overlaps L2 latency w/ WMMA).
struct Frag { V16 a0, a1, b[4]; };

__device__ __forceinline__ void load_frags(Frag& f, const u16* __restrict__ arow0,
                                           const u16* __restrict__ arow1,
                                           const u16* __restrict__ Wp, int Nout,
                                           int n0, int r, int half, int k0) {
  f.a0.u[0] = *(const uint4*)(arow0 + k0 + half * 16);
  f.a0.u[1] = *(const uint4*)(arow0 + k0 + half * 16 + 8);
  f.a1.u[0] = *(const uint4*)(arow1 + k0 + half * 16);
  f.a1.u[1] = *(const uint4*)(arow1 + k0 + half * 16 + 8);
  const size_t kb = (size_t)((k0 >> 4) + half);
  const uint4* bp = (const uint4*)(Wp + (kb * Nout + n0 + r) * 16);
#pragma unroll
  for (int t = 0; t < 4; ++t) {
    f.b[t].u[0] = bp[t * 32];
    f.b[t].u[1] = bp[t * 32 + 1];
  }
}

__device__ __forceinline__ void mma_step(v8f* acc, const Frag& f) {
#pragma unroll
  for (int t = 0; t < 4; ++t) {
    acc[t] = __builtin_amdgcn_wmma_f32_16x16x32_bf16(
        false, f.a0.v, false, f.b[t].v, (short)0, acc[t], false, false);
    acc[4 + t] = __builtin_amdgcn_wmma_f32_16x16x32_bf16(
        false, f.a1.v, false, f.b[t].v, (short)0, acc[4 + t], false, false);
  }
}

__global__ __launch_bounds__(256) void k_wmma_gemm(
    const u16* __restrict__ A, const u16* __restrict__ Wp,
    const float* __restrict__ bias, float* __restrict__ outF,
    u16* __restrict__ outB, int M, int K, int Nout, int relu) {
  const int wave = threadIdx.x >> 5;
  const int lane = threadIdx.x & 31;
  const int wps  = Nout >> 6;          // waves per 32-row strip (64 cols each)
  const int spb  = 8 / wps;            // strips per 256-thread block
  const int strip = blockIdx.x * spb + wave / wps;
  const int m0 = strip << 5;           // 32 rows per strip
  if (m0 >= M) return;
  const int n0   = (wave % wps) << 6;
  const int r    = lane & 15;
  const int half = lane >> 4;

  const u16* arow0 = A + (size_t)(m0 + r) * K;
  const u16* arow1 = A + (size_t)(m0 + 16 + r) * K;

  v8f acc[8] = {v8f{}, v8f{}, v8f{}, v8f{}, v8f{}, v8f{}, v8f{}, v8f{}};

  Frag f0, f1;
  load_frags(f0, arow0, arow1, Wp, Nout, n0, r, half, 0);
  for (int k0 = 0; k0 < K; k0 += 64) {           // K % 64 == 0 here
    if (k0 + 128 < K) {
      __builtin_prefetch(arow0 + k0 + 128, 0, 3);
      __builtin_prefetch(arow1 + k0 + 128, 0, 3);
    }
    if (k0 + 32 < K) load_frags(f1, arow0, arow1, Wp, Nout, n0, r, half, k0 + 32);
    mma_step(acc, f0);                            // overlaps with f1 loads
    if (k0 + 64 < K) load_frags(f0, arow0, arow1, Wp, Nout, n0, r, half, k0 + 64);
    mma_step(acc, f1);                            // overlaps with f0 loads
  }

#pragma unroll
  for (int mt = 0; mt < 2; ++mt) {
#pragma unroll
    for (int t = 0; t < 4; ++t) {
      const int col = n0 + t * 16 + r;
      const float bv = bias ? bias[col] : 0.0f;
      const v8f a = acc[mt * 4 + t];
#pragma unroll
      for (int j = 0; j < 8; ++j) {     // D: VGPR j -> row j (+8 for hi half)
        const int row = m0 + mt * 16 + half * 8 + j;
        float v = a[j] + bv;
        if (relu) v = fmaxf(v, 0.0f);
        const size_t idx = (size_t)row * Nout + col;
        if (outF) outF[idx] = v;
        if (outB) outB[idx] = f2bf(v);
      }
    }
  }
}

// --------------------- per-node alpha_src/alpha_dst dots + init of m/z/accum
__global__ void k_alpha_init(const float* __restrict__ h,
                             const float* __restrict__ a_s,
                             const float* __restrict__ a_d,
                             float* alphaS, float* alphaD, unsigned int* mU,
                             float* z, float* outAcc, int N, int H) {
  const int wave = threadIdx.x >> 5, lane = threadIdx.x & 31;
  const int n = blockIdx.x * 8 + wave;
  if (n >= N) return;
  const float* hr = h + (size_t)n * H;
  float ss = 0.f, sd = 0.f;
  for (int c = lane; c < H; c += 32) {
    float v = hr[c];
    ss += v * a_s[c];
    sd += v * a_d[c];
    outAcc[(size_t)n * H + c] = 0.0f;
  }
  for (int o = 16; o > 0; o >>= 1) { ss += __shfl_xor(ss, o); sd += __shfl_xor(sd, o); }
  if (lane == 0) {
    alphaS[n] = ss;
    alphaD[n] = sd;
    mU[n] = f2ord(-3.0e38f);
    z[n] = 0.0f;
  }
}

__device__ __forceinline__ void edge_sd(const long long* __restrict__ ei,
                                        long long E, long long i,
                                        long long& s, long long& d) {
  if (i < E) { s = ei[i]; d = ei[E + i]; }
  else       { s = i - E; d = i - E; }          // self loops
}

__global__ void k_edge_max(const long long* __restrict__ ei, long long E,
                           long long nE, const float* __restrict__ aS,
                           const float* __restrict__ aD, float* ebuf,
                           unsigned int* mU) {
  long long i = (long long)blockIdx.x * blockDim.x + threadIdx.x;
  if (i >= nE) return;
  long long s, d; edge_sd(ei, E, i, s, d);
  float e = aS[s] + aD[d];
  e = (e >= 0.f) ? e : 0.2f * e;                // leaky relu
  ebuf[i] = e;
  atomicMax(&mU[d], f2ord(e));
}

__global__ void k_edge_sum(const long long* __restrict__ ei, long long E,
                           long long nE, float* ebuf,
                           const unsigned int* __restrict__ mU, float* z) {
  long long i = (long long)blockIdx.x * blockDim.x + threadIdx.x;
  if (i >= nE) return;
  long long s, d; edge_sd(ei, E, i, s, d);
  float ex = __expf(ebuf[i] - ord2f(mU[d]));
  ebuf[i] = ex;
  atomicAdd(&z[d], ex);
}

__global__ void k_edge_aggr(const long long* __restrict__ ei, long long E,
                            long long nE, const float* __restrict__ ebuf,
                            const float* __restrict__ z,
                            const float* __restrict__ h, float* out, int H) {
  const int wave = threadIdx.x >> 5, lane = threadIdx.x & 31;
  long long i = (long long)blockIdx.x * 8 + wave;
  if (i >= nE) return;
  long long s, d; edge_sd(ei, E, i, s, d);
  const float alpha = ebuf[i] / z[d];
  const float* hs = h + (size_t)s * H;
  float* od = out + (size_t)d * H;
  for (int c = lane; c < H; c += 32)
    atomicAdd(&od[c], alpha * hs[c]);
}

__global__ void k_finish(const float* __restrict__ in,
                         const float* __restrict__ bias, float* outF,
                         u16* outB, long long total, int H, int relu) {
  long long i = (long long)blockIdx.x * blockDim.x + threadIdx.x;
  const long long stride = (long long)gridDim.x * blockDim.x;
  for (; i < total; i += stride) {
    float v = in[i] + bias[i % H];
    if (relu) v = fmaxf(v, 0.0f);
    if (outF) outF[i] = v;
    if (outB) outB[i] = f2bf(v);
  }
}

__global__ void k_score(const float* __restrict__ h,
                        const long long* __restrict__ pr, long long E2,
                        float* __restrict__ out) {
  const int wave = threadIdx.x >> 5, lane = threadIdx.x & 31;
  long long i = (long long)blockIdx.x * 8 + wave;
  if (i >= E2) return;
  const long long s = pr[i], d = pr[E2 + i];
  const float* hs = h + (size_t)s * 128;
  const float* hd = h + (size_t)d * 128;
  float acc = 0.f;
#pragma unroll
  for (int j = 0; j < 4; ++j) acc += hs[lane + 32 * j] * hd[lane + 32 * j];
  for (int o = 16; o > 0; o >>= 1) acc += __shfl_xor(acc, o);
  if (lane == 0) out[i] = 1.0f / (1.0f + __expf(-acc));
}

extern "C" void kernel_launch(void* const* d_in, const int* in_sizes, int n_in,
                              void* d_out, int out_size, void* d_ws,
                              size_t ws_size, hipStream_t stream) {
  const float* x    = (const float*)d_in[0];
  const float* linW = (const float*)d_in[1];
  const float* linB = (const float*)d_in[2];
  const float* W1   = (const float*)d_in[3];
  const float* as1  = (const float*)d_in[4];
  const float* ad1  = (const float*)d_in[5];
  const float* b1   = (const float*)d_in[6];
  const float* W2   = (const float*)d_in[7];
  const float* as2  = (const float*)d_in[8];
  const float* ad2  = (const float*)d_in[9];
  const float* b2   = (const float*)d_in[10];
  const long long* ei = (const long long*)d_in[11];  // edge_index int64 [2,E]
  const long long* pr = (const long long*)d_in[12];  // edge       int64 [2,E2]
  float* out = (float*)d_out;

  const int FIN = 512, H = 256, H2 = 128;
  const int N = in_sizes[0] / FIN;                   // 100000, multiple of 32
  const long long E  = in_sizes[11] / 2;
  const long long E2 = in_sizes[12] / 2;
  const long long nE = E + N;

  // ---- workspace carve-up (reuses xb region as the aggregation accumulator)
  char* p = (char*)d_ws;
  u16*   xb   = (u16*)p;
  float* acc3 = (float*)p;            p += (size_t)N * FIN * 2;   // == N*H*4
  u16*   hbf  = (u16*)p;              p += (size_t)N * H * 2;
  float* hf   = (float*)p;            p += (size_t)N * H * 4;
  u16*   Wp0  = (u16*)p;              p += (size_t)(FIN / 16) * H * 16 * 2;
  u16*   Wp1  = (u16*)p;              p += (size_t)(H / 16) * H * 16 * 2;
  u16*   Wp2  = (u16*)p;              p += (size_t)(H / 16) * H2 * 16 * 2;
  float* alphaS = (float*)p;          p += (size_t)N * 4;
  float* alphaD = (float*)p;          p += (size_t)N * 4;
  unsigned int* mU = (unsigned int*)p; p += (size_t)N * 4;
  float* zz   = (float*)p;            p += (size_t)N * 4;
  float* ebuf = (float*)p;            p += (size_t)nE * 4;
  (void)ws_size; (void)n_in; (void)out_size;

  auto gemmGrid = [](int M, int Nout) {
    int strips = (M + 31) / 32;
    int spb = 8 / (Nout >> 6);
    return (strips + spb - 1) / spb;
  };
  const int eThreads = (int)((nE + 255) / 256);
  const int eWaves   = (int)((nE + 7) / 8);

  // h0 = relu(x @ linW + linB), stored bf16
  k_cast_bf16<<<2048, 256, 0, stream>>>(x, xb, (long long)N * FIN / 4);
  k_repack_w<<<((FIN / 16) * H + 255) / 256, 256, 0, stream>>>(linW, Wp0, FIN, H);
  k_repack_w<<<((H / 16) * H + 255) / 256, 256, 0, stream>>>(W1, Wp1, H, H);
  k_repack_w<<<((H / 16) * H2 + 255) / 256, 256, 0, stream>>>(W2, Wp2, H, H2);
  k_wmma_gemm<<<gemmGrid(N, H), 256, 0, stream>>>(xb, Wp0, linB, nullptr, hbf,
                                                  N, FIN, H, 1);

  // ---- GAT layer 1 (H -> H) ----
  k_wmma_gemm<<<gemmGrid(N, H), 256, 0, stream>>>(hbf, Wp1, nullptr, hf,
                                                  nullptr, N, H, H, 0);
  k_alpha_init<<<(N + 7) / 8, 256, 0, stream>>>(hf, as1, ad1, alphaS, alphaD,
                                                mU, zz, acc3, N, H);
  k_edge_max<<<eThreads, 256, 0, stream>>>(ei, E, nE, alphaS, alphaD, ebuf, mU);
  k_edge_sum<<<eThreads, 256, 0, stream>>>(ei, E, nE, ebuf, mU, zz);
  k_edge_aggr<<<eWaves, 256, 0, stream>>>(ei, E, nE, ebuf, zz, hf, acc3, H);
  k_finish<<<4096, 256, 0, stream>>>(acc3, b1, nullptr, hbf, (long long)N * H,
                                     H, 1);

  // ---- GAT layer 2 (H -> H2) ----
  k_wmma_gemm<<<gemmGrid(N, H2), 256, 0, stream>>>(hbf, Wp2, nullptr, hf,
                                                   nullptr, N, H, H2, 0);
  k_alpha_init<<<(N + 7) / 8, 256, 0, stream>>>(hf, as2, ad2, alphaS, alphaD,
                                                mU, zz, acc3, N, H2);
  k_edge_max<<<eThreads, 256, 0, stream>>>(ei, E, nE, alphaS, alphaD, ebuf, mU);
  k_edge_sum<<<eThreads, 256, 0, stream>>>(ei, E, nE, ebuf, mU, zz);
  k_edge_aggr<<<eWaves, 256, 0, stream>>>(ei, E, nE, ebuf, zz, hf, acc3, H2);
  k_finish<<<4096, 256, 0, stream>>>(acc3, b2, acc3, nullptr,
                                     (long long)N * H2, H2, 0);

  // scores = sigmoid(sum(h2[e0] * h2[e1]))
  k_score<<<(int)((E2 + 7) / 8), 256, 0, stream>>>(acc3, pr, E2, out);
}
